// TimeSeriesGNN_31525059953010
// MI455X (gfx1250) — matrix operand order repaired and checked by reading, not compiled, not run on hardware
//
#include <hip/hip_runtime.h>
#include <cstdint>
#include <cstddef>

#define NNODES 32768
#define FIN    32
#define HCC    64
#define NHEADS 4
#define TSTEPS 512
#define NGRAPH 64
#define OUTDIM 24

typedef __attribute__((ext_vector_type(16))) _Float16 v16h;
typedef __attribute__((ext_vector_type(8)))  _Float16 v8h;
typedef __attribute__((ext_vector_type(8)))  float    v8f;

// ---------- helpers ----------
__device__ __forceinline__ unsigned fenc(float f) {
  unsigned u = __float_as_uint(f);
  return (u & 0x80000000u) ? ~u : (u | 0x80000000u);
}
__device__ __forceinline__ float fdec(unsigned k) {
  unsigned u = (k & 0x80000000u) ? (k ^ 0x80000000u) : ~k;
  return __uint_as_float(u);
}

__global__ void fill_u32_kernel(unsigned* p, unsigned v, size_t n) {
  size_t i = (size_t)blockIdx.x * blockDim.x + threadIdx.x;
  size_t stride = (size_t)gridDim.x * blockDim.x;
  for (; i < n; i += stride) p[i] = v;
}

// ---------- f32 -> f16 row-major convert ----------
__global__ void cvt_f16_kernel(const float* __restrict__ in, _Float16* __restrict__ out,
                               size_t n) {
  size_t i = (size_t)blockIdx.x * blockDim.x + threadIdx.x;
  size_t stride = (size_t)gridDim.x * blockDim.x;
  for (; i < n; i += stride) out[i] = (_Float16)in[i];
}

// ---------- swizzle B[K,Nc] (f32 row-major) into per-lane WMMA fragments ----
// Bsw element index = ((kb*tilesN + tn)*32 + lane)*16 + j
//   K-row = kb*32 + (lane>=16 ? 8 : 0) + (j&7) + (j>=8 ? 16 : 0) ; col = tn*16 + (lane&15)
__global__ void swizzleB_kernel(const float* __restrict__ B, _Float16* __restrict__ Bsw,
                                int K, int Nc) {
  int tilesN = Nc >> 4;
  size_t total = (size_t)K * Nc;
  size_t tid = (size_t)blockIdx.x * blockDim.x + threadIdx.x;
  size_t stride = (size_t)gridDim.x * blockDim.x;
  for (; tid < total; tid += stride) {
    int j    = (int)(tid & 15);
    int lane = (int)((tid >> 4) & 31);
    int tn   = (int)((tid >> 9) % tilesN);
    int kb   = (int)(tid / ((size_t)tilesN << 9));
    int krow = (kb << 5) + ((lane >> 4) << 3) + (j & 7) + ((j >> 3) << 4);
    int col  = (tn << 4) + (lane & 15);
    Bsw[tid] = (_Float16)B[(size_t)krow * Nc + col];
  }
}

// ---------- dense GEMM via f16 WMMA, pre-swizzled B, f16 A ----------
// C[M,Nc] = Ah[M,K] @ B, one wave computes a 16x64 strip (4 N-tiles).
// Requires K % 32 == 0, Nc % 64 == 0, M % 16 == 0.
__global__ __launch_bounds__(256)
void gemm_f16sw_kernel(const _Float16* __restrict__ Ah,
                       const _Float16* __restrict__ Bsw,
                       float* __restrict__ C, int M, int K, int Nc) {
  int lane = threadIdx.x & 31;
  int wave = blockIdx.x * (blockDim.x >> 5) + (threadIdx.x >> 5);
  int tilesN = Nc >> 4;
  int groupsN = tilesN >> 2;              // 4 N-tiles per wave
  int tM = wave / groupsN;
  int tNg = wave - tM * groupsN;
  if (tM * 16 >= M) return;

  int mrow = tM * 16 + (lane & 15);
  int kgA = (lane >> 4) << 3;             // lanes 16-31: K+8 half-group
  v8f c0 = {}, c1 = {}, c2 = {}, c3 = {};
  int ksteps = K >> 5;

  for (int kb = 0; kb < ksteps; ++kb) {
    // A fragment: two contiguous 8-half (16B) chunks
    const _Float16* ap = Ah + (size_t)mrow * K + (kb << 5) + kgA;
    v8h alo = *(const v8h*)ap;
    v8h ahi = *(const v8h*)(ap + 16);
    v16h a = __builtin_shufflevector(alo, ahi, 0, 1, 2, 3, 4, 5, 6, 7,
                                     8, 9, 10, 11, 12, 13, 14, 15);
    // B fragments: one contiguous 32B load per N-tile
    const v16h* bp = (const v16h*)Bsw +
                     ((size_t)kb * tilesN + ((size_t)tNg << 2)) * 32 + lane;
    v16h b0 = bp[0];
    v16h b1 = bp[32];
    v16h b2 = bp[64];
    v16h b3 = bp[96];
    c0 = __builtin_amdgcn_wmma_f32_16x16x32_f16(false, a, false, b0, (short)0, c0, false, false);
    c1 = __builtin_amdgcn_wmma_f32_16x16x32_f16(false, a, false, b1, (short)0, c1, false, false);
    c2 = __builtin_amdgcn_wmma_f32_16x16x32_f16(false, a, false, b2, (short)0, c2, false, false);
    c3 = __builtin_amdgcn_wmma_f32_16x16x32_f16(false, a, false, b3, (short)0, c3, false, false);
  }

  int crow = tM * 16 + ((lane >> 4) << 3);    // lanes 16-31 -> M+8
  int ncol0 = (tNg << 6) + (lane & 15);
#pragma unroll
  for (int g = 0; g < 8; ++g) {
    float* cr = C + (size_t)(crow + g) * Nc + ncol0;
    cr[0]  = c0[g];
    cr[16] = c1[g];
    cr[32] = c2[g];
    cr[48] = c3[g];
  }
}

// ---------- attention scores: as[n,h] = <h[n,h,:], a_src[h,:]> ----------
__global__ void attn_scores_kernel(const float* __restrict__ h,
                                   const float* __restrict__ asrc,
                                   const float* __restrict__ adst,
                                   float* __restrict__ as_,
                                   float* __restrict__ ad_,
                                   int n_nodes, int H, int C) {
  int tid = blockIdx.x * blockDim.x + threadIdx.x;
  if (tid >= n_nodes * H) return;
  int n0 = tid / H;
  int hh = tid - n0 * H;
  const float* hr = h + (size_t)n0 * H * C + (size_t)hh * C;
  float sa = 0.f, sd = 0.f;
  for (int c = 0; c < C; ++c) {
    float v = hr[c];
    sa += v * asrc[hh * C + c];
    sd += v * adst[hh * C + c];
  }
  as_[tid] = sa;
  ad_[tid] = sd;
}

// ---------- edge pass 1: segment max via encoded-uint atomicMax ----------
__global__ void edge_max_kernel(const int* __restrict__ src,
                                const int* __restrict__ dst,
                                const float* __restrict__ as_,
                                const float* __restrict__ ad_,
                                unsigned* __restrict__ m,
                                int Etot, int E, int H) {
  long long tid = (long long)blockIdx.x * blockDim.x + threadIdx.x;
  long long total = (long long)Etot * H;
  if (tid >= total) return;
  long long i = tid / H;
  int hh = (int)(tid - i * H);
  int sN, dN;
  if (i < E) { sN = src[i]; dN = dst[i]; } else { sN = dN = (int)(i - E); }
  float e = as_[(size_t)sN * H + hh] + ad_[(size_t)dN * H + hh];
  e = e > 0.f ? e : 0.2f * e;                 // leaky_relu
  atomicMax(m + (size_t)dN * H + hh, fenc(e));
}

// ---------- edge pass 2: w = exp(e - m[dst]); s[dst] += w ----------
__global__ void edge_expsum_kernel(const int* __restrict__ src,
                                   const int* __restrict__ dst,
                                   const float* __restrict__ as_,
                                   const float* __restrict__ ad_,
                                   const unsigned* __restrict__ m,
                                   float* __restrict__ wbuf,
                                   float* __restrict__ sbuf,
                                   int Etot, int E, int H) {
  long long tid = (long long)blockIdx.x * blockDim.x + threadIdx.x;
  long long total = (long long)Etot * H;
  if (tid >= total) return;
  long long i = tid / H;
  int hh = (int)(tid - i * H);
  int sN, dN;
  if (i < E) { sN = src[i]; dN = dst[i]; } else { sN = dN = (int)(i - E); }
  float e = as_[(size_t)sN * H + hh] + ad_[(size_t)dN * H + hh];
  e = e > 0.f ? e : 0.2f * e;
  float wv = __expf(e - fdec(m[(size_t)dN * H + hh]));
  wbuf[tid] = wv;
  atomicAdd(sbuf + (size_t)dN * H + hh, wv);
}

// ---------- edge pass 3: agg[dst] += (w/s[dst]) * h[src], 4 ch per thread ----
__global__ void edge_agg_kernel(const int* __restrict__ src,
                                const int* __restrict__ dst,
                                const float* __restrict__ wbuf,
                                const float* __restrict__ sbuf,
                                const float* __restrict__ hfeat,
                                float* __restrict__ agg,
                                int Etot, int E, int H, int C) {
  int cpg = C >> 2; // channel groups of 4
  long long tid = (long long)blockIdx.x * blockDim.x + threadIdx.x;
  long long total = (long long)Etot * H * cpg;
  if (tid >= total) return;
  long long eh = tid / cpg;
  int cg = (int)(tid - eh * cpg);
  long long i = eh / H;
  int hh = (int)(eh - i * H);
  int sN, dN;
  if (i < E) { sN = src[i]; dN = dst[i]; } else { sN = dN = (int)(i - E); }
  float coef = wbuf[eh] / (sbuf[(size_t)dN * H + hh] + 1e-16f);
  const float4 v = *(const float4*)(hfeat + (size_t)sN * H * C + (size_t)hh * C + cg * 4);
  float* op = agg + (size_t)dN * H * C + (size_t)hh * C + cg * 4;
  atomicAdd(op + 0, coef * v.x);
  atomicAdd(op + 1, coef * v.y);
  atomicAdd(op + 2, coef * v.z);
  atomicAdd(op + 3, coef * v.w);
}

// ---------- elementwise: x = elu(x + b[col]) in-place (f32 out) ----------
__global__ void elu_bias_kernel(float* __restrict__ xio, const float* __restrict__ b,
                                size_t n, int cols) {
  size_t i = (size_t)blockIdx.x * blockDim.x + threadIdx.x;
  size_t stride = (size_t)gridDim.x * blockDim.x;
  for (; i < n; i += stride) {
    float v = xio[i] + b[i % cols];
    xio[i] = v > 0.f ? v : expm1f(v);
  }
}

// ---------- elementwise: out_f16 = elu(in + b[col]) ----------
__global__ void elu_bias_f16_kernel(const float* __restrict__ in,
                                    const float* __restrict__ b,
                                    _Float16* __restrict__ out,
                                    size_t n, int cols) {
  size_t i = (size_t)blockIdx.x * blockDim.x + threadIdx.x;
  size_t stride = (size_t)gridDim.x * blockDim.x;
  for (; i < n; i += stride) {
    float v = in[i] + b[i % cols];
    out[i] = (_Float16)(v > 0.f ? v : expm1f(v));
  }
}

// ---------- GRU: one workgroup per graph, weights staged in LDS ----------
// x layout: [graph, t, HCC] contiguous. 192 threads: one per gate-row.
__global__ __launch_bounds__(192)
void gru_kernel(const float* __restrict__ x,
                float* __restrict__ yseq,   // [graph, t, HCC] (used if store_seq)
                float* __restrict__ hT,     // [graph, HCC] (used if !store_seq)
                const float* __restrict__ Wih, const float* __restrict__ Whh,
                const float* __restrict__ bih, const float* __restrict__ bhh,
                int store_seq) {
  extern __shared__ float smem[];
  float* wih = smem;              // 192*64
  float* whh = wih + 192 * 64;    // 192*64
  float* xb  = whh + 192 * 64;    // 64
  float* hb  = xb + 64;           // 64
  float* gib = hb + 64;           // 192
  float* ghb = gib + 192;         // 192

  int g = blockIdx.x;
  int t = threadIdx.x;            // 0..191

  for (int idx = t; idx < 192 * 64; idx += 192) {
    wih[idx] = Wih[idx];
    whh[idx] = Whh[idx];
  }
  if (t < 64) hb[t] = 0.f;
  float bi = bih[t], bh = bhh[t];
  const float* xg = x + (size_t)g * TSTEPS * HCC;
  float* yg = yseq + (size_t)g * TSTEPS * HCC;
  __syncthreads();

  const float4* wr = (const float4*)(wih + t * 64);
  const float4* vr = (const float4*)(whh + t * 64);
  const float4* xr = (const float4*)xb;
  const float4* hr = (const float4*)hb;

  for (int step = 0; step < TSTEPS; ++step) {
    if (t < 64) xb[t] = xg[(size_t)step * HCC + t];
    __syncthreads();                 // xb ready, previous h update visible

    float accI = bi, accH = bh;
#pragma unroll
    for (int k = 0; k < 16; ++k) {
      float4 a = wr[k], xv = xr[k];
      accI += a.x * xv.x + a.y * xv.y + a.z * xv.z + a.w * xv.w;
      float4 b2 = vr[k], hv = hr[k];
      accH += b2.x * hv.x + b2.y * hv.y + b2.z * hv.z + b2.w * hv.w;
    }
    gib[t] = accI;
    ghb[t] = accH;
    __syncthreads();                 // all dot products done before h update

    if (t < 64) {
      float r = 1.f / (1.f + expf(-(gib[t] + ghb[t])));
      float z = 1.f / (1.f + expf(-(gib[64 + t] + ghb[64 + t])));
      float nn = tanhf(gib[128 + t] + r * ghb[128 + t]);
      float hnew = (1.f - z) * nn + z * hb[t];
      hb[t] = hnew;
      if (store_seq) yg[(size_t)step * HCC + t] = hnew;
    }
  }
  __syncthreads();
  if (!store_seq && t < 64) hT[(size_t)g * HCC + t] = hb[t];
}

// ---------- head: out[g,o] = <hT[g,:], Wl[o,:]> + bl[o] ----------
__global__ void head_kernel(const float* __restrict__ hT,
                            const float* __restrict__ Wl,
                            const float* __restrict__ bl,
                            float* __restrict__ out) {
  int tid = blockIdx.x * blockDim.x + threadIdx.x;
  if (tid >= NGRAPH * OUTDIM) return;
  int g = tid / OUTDIM;
  int o = tid - g * OUTDIM;
  float acc = bl[o];
  for (int k = 0; k < HCC; ++k)
    acc += hT[(size_t)g * HCC + k] * Wl[(size_t)o * HCC + k];
  out[tid] = acc;
}

// ---------- launcher ----------
extern "C" void kernel_launch(void* const* d_in, const int* in_sizes, int n_in,
                              void* d_out, int out_size, void* d_ws, size_t ws_size,
                              hipStream_t stream) {
  (void)n_in; (void)out_size; (void)ws_size;
  const float* x    = (const float*)d_in[0];
  const int*   ei   = (const int*)d_in[1];
  // d_in[2] = batch (layout is static: graph = n / T)
  const float* W1   = (const float*)d_in[3];
  const float* aS1  = (const float*)d_in[4];
  const float* aD1  = (const float*)d_in[5];
  const float* b1   = (const float*)d_in[6];
  const float* W2   = (const float*)d_in[7];
  const float* aS2  = (const float*)d_in[8];
  const float* aD2  = (const float*)d_in[9];
  const float* b2   = (const float*)d_in[10];
  const float* Wih0 = (const float*)d_in[11];
  const float* Whh0 = (const float*)d_in[12];
  const float* bih0 = (const float*)d_in[13];
  const float* bhh0 = (const float*)d_in[14];
  const float* Wih1 = (const float*)d_in[15];
  const float* Whh1 = (const float*)d_in[16];
  const float* bih1 = (const float*)d_in[17];
  const float* bhh1 = (const float*)d_in[18];
  const float* Wl   = (const float*)d_in[19];
  const float* bl   = (const float*)d_in[20];

  const int E = in_sizes[1] / 2;
  const int Etot = E + NNODES;
  const int* src = ei;
  const int* dst = ei + E;

  float* ws = (float*)d_ws;
  size_t off = 0;
  float* h1   = ws + off; off += (size_t)NNODES * NHEADS * HCC;   // 32 MB
  float* agg1 = ws + off; off += (size_t)NNODES * NHEADS * HCC;   // 32 MB
  float* h2   = ws + off; off += (size_t)NNODES * HCC;            // 8 MB
  float* agg2 = ws + off; off += (size_t)NNODES * HCC;            // 8 MB
  float* wbuf = ws + off; off += (size_t)Etot * NHEADS;           // edge weights
  float* as1  = ws + off; off += (size_t)NNODES * NHEADS;
  float* ad1  = ws + off; off += (size_t)NNODES * NHEADS;
  float* s1   = ws + off; off += (size_t)NNODES * NHEADS;
  unsigned* m1 = (unsigned*)(ws + off); off += (size_t)NNODES * NHEADS;
  float* as2  = ws + off; off += NNODES;
  float* ad2  = ws + off; off += NNODES;
  float* s2   = ws + off; off += NNODES;
  unsigned* m2 = (unsigned*)(ws + off); off += NNODES;
  float* hT   = ws + off; off += (size_t)NGRAPH * HCC;
  _Float16* xh   = (_Float16*)(ws + off); off += (size_t)NNODES * FIN / 2;  // f16 x
  _Float16* W1sw = (_Float16*)(ws + off); off += (size_t)FIN * NHEADS * HCC / 2;
  _Float16* W2sw = (_Float16*)(ws + off); off += (size_t)NHEADS * HCC * HCC / 2;
  // Aliases into dead h1 region:
  _Float16* act1h = (_Float16*)h1;  // f16 elu(agg1+b1): consumed by GEMM2, before GRU
  float* seq0 = h1;                 // GRU layer-0 output: written after GEMM2 read

  // ===== prepare f16 operands =====
  cvt_f16_kernel<<<512, 256, 0, stream>>>(x, xh, (size_t)NNODES * FIN);
  swizzleB_kernel<<<32, 256, 0, stream>>>(W1, W1sw, FIN, NHEADS * HCC);
  swizzleB_kernel<<<64, 256, 0, stream>>>(W2, W2sw, NHEADS * HCC, HCC);

  // ===== GAT layer 1 =====
  {
    int waves = (NNODES / 16) * ((NHEADS * HCC) / 64);   // 16x64 strip per wave
    gemm_f16sw_kernel<<<(waves + 7) / 8, 256, 0, stream>>>(xh, W1sw, h1,
                                                           NNODES, FIN, NHEADS * HCC);
  }
  {
    int tot = NNODES * NHEADS;
    attn_scores_kernel<<<(tot + 255) / 256, 256, 0, stream>>>(h1, aS1, aD1, as1, ad1,
                                                              NNODES, NHEADS, HCC);
  }
  fill_u32_kernel<<<2048, 256, 0, stream>>>((unsigned*)agg1, 0u, (size_t)NNODES * NHEADS * HCC);
  fill_u32_kernel<<<256, 256, 0, stream>>>((unsigned*)s1, 0u, (size_t)NNODES * NHEADS);
  fill_u32_kernel<<<256, 256, 0, stream>>>(m1, 0x007FFFFFu, (size_t)NNODES * NHEADS); // enc(-inf)
  {
    long long tot = (long long)Etot * NHEADS;
    int blocks = (int)((tot + 255) / 256);
    edge_max_kernel<<<blocks, 256, 0, stream>>>(src, dst, as1, ad1, m1, Etot, E, NHEADS);
    edge_expsum_kernel<<<blocks, 256, 0, stream>>>(src, dst, as1, ad1, m1, wbuf, s1,
                                                   Etot, E, NHEADS);
  }
  {
    long long tot = (long long)Etot * NHEADS * (HCC / 4);
    edge_agg_kernel<<<(int)((tot + 255) / 256), 256, 0, stream>>>(src, dst, wbuf, s1,
                                                                  h1, agg1, Etot, E,
                                                                  NHEADS, HCC);
  }
  // f16 activation for the GEMM2 A-operand (overwrites dead h1 region)
  elu_bias_f16_kernel<<<2048, 256, 0, stream>>>(agg1, b1, act1h,
                                                (size_t)NNODES * NHEADS * HCC,
                                                NHEADS * HCC);

  // ===== GAT layer 2 =====
  {
    int waves = (NNODES / 16) * (HCC / 64);
    gemm_f16sw_kernel<<<(waves + 7) / 8, 256, 0, stream>>>(act1h, W2sw, h2,
                                                           NNODES, NHEADS * HCC, HCC);
  }
  attn_scores_kernel<<<(NNODES + 255) / 256, 256, 0, stream>>>(h2, aS2, aD2, as2, ad2,
                                                               NNODES, 1, HCC);
  fill_u32_kernel<<<1024, 256, 0, stream>>>((unsigned*)agg2, 0u, (size_t)NNODES * HCC);
  fill_u32_kernel<<<128, 256, 0, stream>>>((unsigned*)s2, 0u, (size_t)NNODES);
  fill_u32_kernel<<<128, 256, 0, stream>>>(m2, 0x007FFFFFu, (size_t)NNODES);
  {
    int blocks = (Etot + 255) / 256;
    edge_max_kernel<<<blocks, 256, 0, stream>>>(src, dst, as2, ad2, m2, Etot, E, 1);
    edge_expsum_kernel<<<blocks, 256, 0, stream>>>(src, dst, as2, ad2, m2, wbuf, s2,
                                                   Etot, E, 1);
  }
  {
    long long tot = (long long)Etot * (HCC / 4);
    edge_agg_kernel<<<(int)((tot + 255) / 256), 256, 0, stream>>>(src, dst, wbuf, s2,
                                                                  h2, agg2, Etot, E,
                                                                  1, HCC);
  }
  elu_bias_kernel<<<1024, 256, 0, stream>>>(agg2, b2, (size_t)NNODES * HCC, HCC);

  // ===== GRU (2 layers), one workgroup per graph =====
  size_t shmem = (size_t)(2 * 192 * 64 + 64 + 64 + 192 + 192) * sizeof(float);
  gru_kernel<<<NGRAPH, 192, shmem, stream>>>(agg2, seq0, hT, Wih0, Whh0, bih0, bhh0, 1);
  gru_kernel<<<NGRAPH, 192, shmem, stream>>>(seq0, seq0, hT, Wih1, Whh1, bih1, bhh1, 0);

  // ===== Linear head =====
  head_kernel<<<(NGRAPH * OUTDIM + 255) / 256, 256, 0, stream>>>(hT, Wl, bl, (float*)d_out);
}
